// ERGCNConv_83056077570511
// MI455X (gfx1250) — matrix operand order repaired and checked by reading, not compiled, not run on hardware
//
#include <hip/hip_runtime.h>
#include <hip/hip_bf16.h>

// ---------------------------------------------------------------------------
// ERGCN conv for MI455X (gfx1250, wave32).
//   K1 init:  zero counts/agg, convert feats fp32 -> bf16
//   K2 count: counts[v*16+t] += 1 per edge (float atomics)
//   K3 edge:  per-etype workgroups; ballot-compact edges into 16-row groups,
//             bf16 WMMA (16x16x32) against LDS-resident W_t, scale by 1/count,
//             atomic-scatter into agg
//   K4 final: VALU node transform + agg + active-relation edge biases + ReLU
// ---------------------------------------------------------------------------

typedef __attribute__((ext_vector_type(16))) __bf16 v16bf;
typedef __attribute__((ext_vector_type(8)))  float  v8f;

#define D 128
#define T_REL 16
#define EDGE_CHUNK 8192
#define WAVE_CHUNK 1024   // EDGE_CHUNK / 8 waves

union FragU { unsigned int u[8]; v16bf v; };

static __device__ __forceinline__ void lds_fence() {
    asm volatile("s_wait_dscnt 0" ::: "memory");
    __builtin_amdgcn_wave_barrier();
}

// ---------------------------------------------------------------------------
__global__ void k_init(const float* __restrict__ feats,
                       unsigned short* __restrict__ featsBF,
                       float* __restrict__ agg,
                       float* __restrict__ counts,
                       int nFeat, int nCnt) {
    int i = blockIdx.x * blockDim.x + threadIdx.x;
    if (i < nFeat) {
        agg[i] = 0.0f;
        __bf16 h = (__bf16)feats[i];
        featsBF[i] = __builtin_bit_cast(unsigned short, h);
    }
    if (i < nCnt) counts[i] = 0.0f;
}

// ---------------------------------------------------------------------------
__global__ void k_count(const int* __restrict__ dst,
                        const int* __restrict__ etypes,
                        float* __restrict__ counts, int E) {
    int e = blockIdx.x * blockDim.x + threadIdx.x;
    if (e < E) atomicAdd(&counts[dst[e] * T_REL + etypes[e]], 1.0f);
}

// ---------------------------------------------------------------------------
// Process one group of up to 16 edges (all same etype) with WMMA.
static __device__ __forceinline__ void process_group(
    const int* __restrict__ pend, int cnt, int lane, int t,
    const unsigned short* __restrict__ featsBF,
    const int* __restrict__ src, const int* __restrict__ dst,
    const float* __restrict__ counts,
    const unsigned int* __restrict__ sW,
    float* __restrict__ agg) {

    int   edgeSrc = 0, edgeDst = 0;
    float edgeInv = 0.0f;                       // padding rows contribute 0
    if (lane < cnt) {
        int e   = pend[lane];
        edgeSrc = src[e];
        edgeDst = dst[e];
        edgeInv = 1.0f / counts[edgeDst * T_REL + t];
    }

    const bool hi = (lane >= 16);
    const int  m  = lane & 15;                  // A-matrix row this lane serves
    const int  srcm = __shfl(edgeSrc, m);
    const unsigned short* frow = featsBF + (size_t)srcm * D;

    // ---- A fragments: 16x32 bf16, lanes 0-15: K 0..7/16..23, hi: +8 -------
    FragU A[4];
#pragma unroll
    for (int kk = 0; kk < 4; ++kk) {
        int kb = kk * 32 + (hi ? 8 : 0);
#pragma unroll
        for (int v = 0; v < 4; ++v) {
            A[kk].u[v]     = *(const unsigned int*)(frow + kb + 2 * v);
            A[kk].u[v + 4] = *(const unsigned int*)(frow + kb + 16 + 2 * v);
        }
    }

    const int n = lane & 15;                    // output column within f-tile
#pragma unroll
    for (int ft = 0; ft < 8; ++ft) {
        v8f acc = {0.f, 0.f, 0.f, 0.f, 0.f, 0.f, 0.f, 0.f};
#pragma unroll
        for (int kk = 0; kk < 4; ++kk) {
            // B fragment: 32x16 bf16 from LDS (k-pair packed dwords)
            FragU B;
            int kp0 = kk * 16 + (hi ? 4 : 0);
            int col = ft * 16 + n;
#pragma unroll
            for (int v = 0; v < 8; ++v)
                B.u[v] = sW[(kp0 + v + (v >= 4 ? 4 : 0)) * D + col];
            acc = __builtin_amdgcn_wmma_f32_16x16x32_bf16(
                false, A[kk].v, false, B.v, (short)0, acc, false, false);
        }
        // D layout: VGPR r holds row m = r (+8 for hi lanes), col = lane&15
#pragma unroll
        for (int r = 0; r < 8; ++r) {
            int   mr = r + (hi ? 8 : 0);
            float iv = __shfl(edgeInv, mr);
            int   dv = __shfl(edgeDst, mr);
            atomicAdd(&agg[(size_t)dv * D + ft * 16 + n], acc[r] * iv);
        }
    }
}

// ---------------------------------------------------------------------------
__global__ __launch_bounds__(256)
void k_edge(const float* __restrict__ W_edge,
            const unsigned short* __restrict__ featsBF,
            const int* __restrict__ src, const int* __restrict__ dst,
            const int* __restrict__ etypes,
            const float* __restrict__ counts,
            float* __restrict__ agg, int E) {
    const int t = blockIdx.y;

    // ---- stage W_t into LDS as bf16, k-pair packed: sW[kp*128+n] = {k0,k1}
    __shared__ unsigned int sW[64 * D];
    const float* Wg = W_edge + (size_t)t * D * D;
    for (int idx = threadIdx.x; idx < 64 * D; idx += blockDim.x) {
        int kp = idx >> 7, col = idx & 127;
        __bf16 w0 = (__bf16)Wg[(2 * kp) * D + col];
        __bf16 w1 = (__bf16)Wg[(2 * kp + 1) * D + col];
        unsigned int p =
            ((unsigned int)__builtin_bit_cast(unsigned short, w1) << 16) |
             (unsigned int)__builtin_bit_cast(unsigned short, w0);
        sW[idx] = p;
    }
    __syncthreads();

    __shared__ int sPend[8][48];
    const int wave = threadIdx.x >> 5;
    const int lane = threadIdx.x & 31;
    int* pend = sPend[wave];

    int waveStart = blockIdx.x * EDGE_CHUNK + wave * WAVE_CHUNK;
    int waveEnd   = min(waveStart + WAVE_CHUNK, E);

    int npend = 0;
    for (int base = waveStart; base < waveEnd; base += 32) {
        int  e = base + lane;
        bool match = (e < waveEnd) && (etypes[e] == t);
        unsigned int mask = (unsigned int)__ballot(match);
        int prefix = __popc(mask & ((1u << lane) - 1u));
        if (match) pend[npend + prefix] = e;
        npend += __popc(mask);
        lds_fence();
        while (npend >= 16) {
            process_group(pend, 16, lane, t, featsBF, src, dst, counts, sW, agg);
            int rem = npend - 16;
            int vtmp = 0;
            if (lane < rem) vtmp = pend[16 + lane];
            lds_fence();
            if (lane < rem) pend[lane] = vtmp;
            lds_fence();
            npend = rem;
        }
    }
    if (npend > 0)
        process_group(pend, npend, lane, t, featsBF, src, dst, counts, sW, agg);
}

// ---------------------------------------------------------------------------
__global__ void k_final(const float* __restrict__ feats,
                        const float* __restrict__ W_node,
                        const float* __restrict__ b_node,
                        const float* __restrict__ b_edge,
                        const int* __restrict__ ntypes,
                        const float* __restrict__ counts,
                        const float* __restrict__ agg,
                        float* __restrict__ out) {
    const int v = blockIdx.x;
    const int f = threadIdx.x;
    __shared__ float sf[D];
    sf[f] = feats[(size_t)v * D + f];
    __syncthreads();

    const int tn = ntypes[v];
    const float* W = W_node + (size_t)tn * D * D;
    float sum = b_node[tn * D + f];
#pragma unroll 8
    for (int d = 0; d < D; ++d)
        sum = fmaf(sf[d], W[d * D + f], sum);

    sum += agg[(size_t)v * D + f];

    const float* c = counts + (size_t)v * T_REL;
#pragma unroll
    for (int t = 0; t < T_REL; ++t)
        if (c[t] > 0.0f) sum += b_edge[t * D + f];

    out[(size_t)v * D + f] = fmaxf(sum, 0.0f);
}

// ---------------------------------------------------------------------------
extern "C" void kernel_launch(void* const* d_in, const int* in_sizes, int n_in,
                              void* d_out, int out_size, void* d_ws, size_t ws_size,
                              hipStream_t stream) {
    const float* feats  = (const float*)d_in[0];
    const float* W_edge = (const float*)d_in[1];
    const float* b_edge = (const float*)d_in[2];
    const float* W_node = (const float*)d_in[3];
    const float* b_node = (const float*)d_in[4];
    const int*   src    = (const int*)d_in[5];
    const int*   dst    = (const int*)d_in[6];
    const int*   ntypes = (const int*)d_in[7];
    const int*   etypes = (const int*)d_in[8];

    const int N = in_sizes[0] / D;
    const int E = in_sizes[5];

    float* out = (float*)d_out;

    // workspace layout: counts [N*16 f32] | agg [N*128 f32] | featsBF [N*128 bf16]
    char* w = (char*)d_ws;
    float*          counts = (float*)w;
    float*          agg    = (float*)(w + (size_t)N * T_REL * 4);
    unsigned short* fbf    = (unsigned short*)(w + (size_t)N * T_REL * 4 +
                                                   (size_t)N * D * 4);

    const int nFeat = N * D;
    const int nCnt  = N * T_REL;

    k_init<<<(nFeat + 255) / 256, 256, 0, stream>>>(feats, fbf, agg, counts,
                                                    nFeat, nCnt);
    k_count<<<(E + 255) / 256, 256, 0, stream>>>(dst, etypes, counts, E);

    dim3 grid((E + EDGE_CHUNK - 1) / EDGE_CHUNK, T_REL);
    k_edge<<<grid, 256, 0, stream>>>(W_edge, fbf, src, dst, etypes, counts,
                                     agg, E);

    k_final<<<N, D, 0, stream>>>(feats, W_node, b_node, b_edge, ntypes, counts,
                                 agg, out);
}
